// GraphSAGEAccessibilitySVIGNN_42777874268502
// MI455X (gfx1250) — compile-verified, matrix operand order checked by reading, not run on hardware
//
#include <hip/hip_runtime.h>
#include <hip/hip_bf16.h>

typedef __bf16 bf16;
typedef __attribute__((ext_vector_type(16))) __bf16 v16bf;
typedef __attribute__((ext_vector_type(8)))  float  v8f;

#define LN_EPS 1e-5f
#define CDIV(a, b) (((a) + (b) - 1) / (b))

__device__ __forceinline__ float bf2f(bf16 v) { return (float)v; }
__device__ __forceinline__ bf16  f2bf(float v) { return (bf16)v; }

// ---------------------------------------------------------------------------
// WMMA fragment helpers (CDNA5 16x16x32 bf16, wave32 layouts per ISA 7.12.2)
//   A 16x32: lane L (L<16) holds row M=L, K in {koff..koff+7, koff+16..koff+23}
//            with koff = (L>=16) ? 8 : 0
//   B 32x16: mirrored with N in place of M; B[k][n] = W[n][k] (W row-major),
//            so a B fragment load == an A fragment load from W at row0=n0.
//   C/D 16x16 f32: lane L holds col N=L&15, VGPR r -> row M=r+(L>=16?8:0)
// ---------------------------------------------------------------------------
__device__ __forceinline__ v16bf load_frag_bf16(const bf16* __restrict__ X, int ld,
                                                int row0, int k0, int lane) {
  const int m = lane & 15;
  const int koff = (lane >> 4) << 3;
  const bf16* p = X + (long)(row0 + m) * ld + k0 + koff;
  v16bf a;
#pragma unroll
  for (int i = 0; i < 8; ++i) a[i] = p[i];
#pragma unroll
  for (int i = 0; i < 8; ++i) a[8 + i] = p[16 + i];
  return a;
}

__device__ __forceinline__ v16bf load_frag_f32_scaled(const float* __restrict__ S, int ld,
                                                      int row0, int k0, int lane, float sc) {
  const int m = lane & 15;
  const int koff = (lane >> 4) << 3;
  const float* p = S + (long)(row0 + m) * ld + k0 + koff;
  v16bf a;
#pragma unroll
  for (int i = 0; i < 8; ++i) a[i] = f2bf(p[i] * sc);
#pragma unroll
  for (int i = 0; i < 8; ++i) a[8 + i] = f2bf(p[16 + i] * sc);
  return a;
}

// ---------------------------------------------------------------------------
// utility kernels
// ---------------------------------------------------------------------------
__global__ void zero_f32(float* __restrict__ p, int n) {
  int i = blockIdx.x * blockDim.x + threadIdx.x;
  if (i < n) p[i] = 0.0f;
}

// one-shot fp32 -> bf16 weight conversion (weights are static per launch)
__global__ void cvt_w_bf16(const float* __restrict__ W, bf16* __restrict__ Wb, int n) {
  int i = blockIdx.x * blockDim.x + threadIdx.x;
  if (i < n) Wb[i] = f2bf(W[i]);
}

__global__ void degree_kernel(const int* __restrict__ dst, float* __restrict__ cnt, int E) {
  int e = blockIdx.x * blockDim.x + threadIdx.x;
  if (e < E) atomicAdd(&cnt[dst[e]], 1.0f);
}

// ---------------------------------------------------------------------------
// stage 1: context gate + softmax attention + input layernorm  ->  x0 bf16 [N,32]
// ---------------------------------------------------------------------------
__global__ void ctx_gate_ln(const float* __restrict__ ACC, const float* __restrict__ CTX,
                            const float* __restrict__ w1, const float* __restrict__ b1,
                            const float* __restrict__ w2, const float* __restrict__ b2,
                            const float* __restrict__ aw, const float* __restrict__ ab,
                            const float* __restrict__ bimp,
                            const float* __restrict__ lng, const float* __restrict__ lnb,
                            bf16* __restrict__ X0, int N) {
  __shared__ float s_w1[32 * 5], s_w2[32 * 32], s_aw[32 * 32];
  __shared__ float s_b1[32], s_b2[32], s_ab[32], s_bi[32], s_g[32], s_b[32];
  for (int t = threadIdx.x; t < 32 * 5; t += blockDim.x) s_w1[t] = w1[t];
  for (int t = threadIdx.x; t < 32 * 32; t += blockDim.x) { s_w2[t] = w2[t]; s_aw[t] = aw[t]; }
  if (threadIdx.x < 32) {
    int j = threadIdx.x;
    s_b1[j] = b1[j]; s_b2[j] = b2[j]; s_ab[j] = ab[j];
    s_bi[j] = bimp[j]; s_g[j] = lng[j]; s_b[j] = lnb[j];
  }
  __syncthreads();
  int i = blockIdx.x * blockDim.x + threadIdx.x;
  if (i >= N) return;

  float c[5];
#pragma unroll
  for (int k = 0; k < 5; ++k) c[k] = CTX[(long)i * 5 + k];

  float h1[32];
  for (int j = 0; j < 32; ++j) {
    float s = s_b1[j];
#pragma unroll
    for (int k = 0; k < 5; ++k) s += s_w1[j * 5 + k] * c[k];
    h1[j] = fmaxf(s, 0.0f);
  }
  float h2[32];
  for (int j = 0; j < 32; ++j) {
    float s = s_b2[j];
    for (int k = 0; k < 32; ++k) s += s_w2[j * 32 + k] * h1[k];
    h2[j] = s;
  }
  float lg[32];
  float mx = -1e30f;
  for (int j = 0; j < 32; ++j) {
    float s = s_ab[j];
    for (int k = 0; k < 32; ++k) s += s_aw[j * 32 + k] * h2[k];
    lg[j] = s;
    mx = fmaxf(mx, s);
  }
  float sum = 0.0f;
  for (int j = 0; j < 32; ++j) { lg[j] = expf(lg[j] - mx); sum += lg[j]; }
  float inv = 1.0f / sum;

  float x[32];
  float mu = 0.0f;
  for (int j = 0; j < 32; ++j) {
    x[j] = ACC[(long)i * 32 + j] * (lg[j] * inv * s_bi[j]);
    mu += x[j];
  }
  mu *= (1.0f / 32.0f);
  float var = 0.0f;
  for (int j = 0; j < 32; ++j) { float d = x[j] - mu; var += d * d; }
  var *= (1.0f / 32.0f);
  float r = rsqrtf(var + LN_EPS);
  for (int j = 0; j < 32; ++j)
    X0[(long)i * 32 + j] = f2bf((x[j] - mu) * r * s_g[j] + s_b[j]);
}

// ---------------------------------------------------------------------------
// WMMA GEMM: Y[N,NOUT] = relu(X[N,KIN] @ W.T + bias), bf16 in/out, f32 accum
// weights pre-converted to bf16; one wave = one 16-row tile
// ---------------------------------------------------------------------------
template <int KIN, int NOUT>
__global__ void gemm_bias_relu(const bf16* __restrict__ X, const bf16* __restrict__ W,
                               const float* __restrict__ bias, bf16* __restrict__ Y, int nrows) {
  const int lane = threadIdx.x & 31;
  const int wave = threadIdx.x >> 5;
  const int tile = blockIdx.x * (blockDim.x >> 5) + wave;
  const int row0 = tile * 16;
  if (row0 >= nrows) return;

  constexpr int KS = KIN / 32;
  v16bf a[KS];
#pragma unroll
  for (int k = 0; k < KS; ++k) a[k] = load_frag_bf16(X, KIN, row0, k * 32, lane);

  const int n = lane & 15;
  const int mbase = row0 + ((lane >> 4) << 3);
#pragma unroll
  for (int nt = 0; nt < NOUT / 16; ++nt) {
    const int n0 = nt * 16;
    const float bv = bias[n0 + n];
    v8f acc;
#pragma unroll
    for (int r = 0; r < 8; ++r) acc[r] = bv;
#pragma unroll
    for (int k = 0; k < KS; ++k) {
      v16bf b = load_frag_bf16(W, KIN, n0, k * 32, lane);
      acc = __builtin_amdgcn_wmma_f32_16x16x32_bf16(false, a[k], false, b, (short)0, acc,
                                                    false, false);
    }
#pragma unroll
    for (int r = 0; r < 8; ++r) {
      float v = fmaxf(acc[r], 0.0f);
      Y[(long)(mbase + r) * NOUT + n0 + n] = f2bf(v);
    }
  }
}

// ---------------------------------------------------------------------------
// edge aggregation: S[dst] += x[src]   (f32 atomics into L2-resident buffer)
// H/16 threads per edge, 16 features each
// ---------------------------------------------------------------------------
template <int H>
__global__ void aggregate(const bf16* __restrict__ X, const int* __restrict__ src,
                          const int* __restrict__ dst, float* __restrict__ S, int E) {
  constexpr int TPE = H / 16;
  int t = blockIdx.x * blockDim.x + threadIdx.x;
  int e = t / TPE;
  int part = t % TPE;
  if (e >= E) return;
  const bf16* xp = X + (long)src[e] * H + part * 16;
  float* sp = S + (long)dst[e] * H + part * 16;
#pragma unroll
  for (int i = 0; i < 16; ++i) atomicAdd(&sp[i], bf2f(xp[i]));
}

// ---------------------------------------------------------------------------
// SAGE combine (WMMA): OUT = (S/cnt) @ WL.T + BL + X @ WR.T   (pre-BN, f32 out)
// ---------------------------------------------------------------------------
template <int HIN, int HOUT>
__global__ void sage_combine(const bf16* __restrict__ X, const float* __restrict__ S,
                             const float* __restrict__ CNT, const bf16* __restrict__ WL,
                             const float* __restrict__ BL, const bf16* __restrict__ WR,
                             float* __restrict__ OUT, int nrows) {
  const int lane = threadIdx.x & 31;
  const int wave = threadIdx.x >> 5;
  const int tile = blockIdx.x * (blockDim.x >> 5) + wave;
  const int row0 = tile * 16;
  if (row0 >= nrows) return;

  constexpr int KS = HIN / 32;
  const int m = lane & 15;
  const float inv = 1.0f / fmaxf(CNT[row0 + m], 1.0f);

  v16bf ax[KS], ag[KS];
#pragma unroll
  for (int k = 0; k < KS; ++k) {
    ax[k] = load_frag_bf16(X, HIN, row0, k * 32, lane);
    ag[k] = load_frag_f32_scaled(S, HIN, row0, k * 32, lane, inv);
  }

  const int mbase = row0 + ((lane >> 4) << 3);
#pragma unroll
  for (int nt = 0; nt < HOUT / 16; ++nt) {
    const int n0 = nt * 16;
    const float bv = BL[n0 + m];
    v8f acc;
#pragma unroll
    for (int r = 0; r < 8; ++r) acc[r] = bv;
#pragma unroll
    for (int k = 0; k < KS; ++k) {
      v16bf b = load_frag_bf16(WR, HIN, n0, k * 32, lane);
      acc = __builtin_amdgcn_wmma_f32_16x16x32_bf16(false, ax[k], false, b, (short)0, acc,
                                                    false, false);
    }
#pragma unroll
    for (int k = 0; k < KS; ++k) {
      v16bf b = load_frag_bf16(WL, HIN, n0, k * 32, lane);
      acc = __builtin_amdgcn_wmma_f32_16x16x32_bf16(false, ag[k], false, b, (short)0, acc,
                                                    false, false);
    }
#pragma unroll
    for (int r = 0; r < 8; ++r) OUT[(long)(mbase + r) * HOUT + n0 + m] = acc[r];
  }
}

// ---------------------------------------------------------------------------
// batchnorm (training-mode stats): two-pass. stat = [sum(H), sumsq(H)]
// ---------------------------------------------------------------------------
template <int H>
__global__ void bn_stats(const float* __restrict__ X, float* __restrict__ stat, int nrows) {
  __shared__ float ls[256], lq[256];
  constexpr int G = 256 / H;
  const int c = threadIdx.x % H;
  const int g = threadIdx.x / H;
  float s = 0.0f, q = 0.0f;
  for (int i = blockIdx.x * G + g; i < nrows; i += gridDim.x * G) {
    float v = X[(long)i * H + c];
    s += v;
    q += v * v;
  }
  ls[threadIdx.x] = s;
  lq[threadIdx.x] = q;
  __syncthreads();
  if (g == 0) {
#pragma unroll
    for (int k = 1; k < G; ++k) { s += ls[k * H + c]; q += lq[k * H + c]; }
    atomicAdd(&stat[c], s);
    atomicAdd(&stat[H + c], q);
  }
}

template <int H>
__global__ void bn_apply_relu(const float* __restrict__ X, const float* __restrict__ stat,
                              const float* __restrict__ G_, const float* __restrict__ B_,
                              bf16* __restrict__ Y, int nrows) {
  int t = blockIdx.x * blockDim.x + threadIdx.x;
  if (t >= nrows * H) return;
  int c = t % H;
  float invn = 1.0f / (float)nrows;
  float mu = stat[c] * invn;
  float var = stat[H + c] * invn - mu * mu;
  float v = (X[t] - mu) * rsqrtf(var + LN_EPS) * G_[c] + B_[c];
  Y[t] = f2bf(fmaxf(v, 0.0f));
}

// ---------------------------------------------------------------------------
// SVI head: sigmoid(relu(x @ W1.T + b1) @ W2.T + b2)  -> f32 out
// ---------------------------------------------------------------------------
__global__ void svi_head(const bf16* __restrict__ X, const float* __restrict__ W1,
                         const float* __restrict__ B1, const float* __restrict__ W2,
                         const float* __restrict__ B2, float* __restrict__ out, int N) {
  __shared__ float s_w1[16 * 32], s_b1[16], s_w2[16];
  for (int t = threadIdx.x; t < 16 * 32; t += blockDim.x) s_w1[t] = W1[t];
  if (threadIdx.x < 16) { s_b1[threadIdx.x] = B1[threadIdx.x]; s_w2[threadIdx.x] = W2[threadIdx.x]; }
  __syncthreads();
  int i = blockIdx.x * blockDim.x + threadIdx.x;
  if (i >= N) return;
  float x[32];
  for (int k = 0; k < 32; ++k) x[k] = bf2f(X[(long)i * 32 + k]);
  float o = B2[0];
  for (int j = 0; j < 16; ++j) {
    float h = s_b1[j];
    for (int k = 0; k < 32; ++k) h += s_w1[j * 32 + k] * x[k];
    o += s_w2[j] * fmaxf(h, 0.0f);
  }
  out[i] = 1.0f / (1.0f + expf(-o));
}

// ---------------------------------------------------------------------------
// host orchestration
// ---------------------------------------------------------------------------
extern "C" void kernel_launch(void* const* d_in, const int* in_sizes, int n_in,
                              void* d_out, int out_size, void* d_ws, size_t ws_size,
                              hipStream_t stream) {
  (void)n_in; (void)out_size; (void)ws_size;

  const float* acc    = (const float*)d_in[0];
  const float* ctx    = (const float*)d_in[1];
  const int*   edge   = (const int*)d_in[2];
  const float* ctx_w1 = (const float*)d_in[3];
  const float* ctx_b1 = (const float*)d_in[4];
  const float* ctx_w2 = (const float*)d_in[5];
  const float* ctx_b2 = (const float*)d_in[6];
  const float* attn_w = (const float*)d_in[7];
  const float* attn_b = (const float*)d_in[8];
  const float* base_i = (const float*)d_in[9];
  const float* ln_g   = (const float*)d_in[10];
  const float* ln_b   = (const float*)d_in[11];
  const float* fe_w1  = (const float*)d_in[12];
  const float* fe_b1  = (const float*)d_in[13];
  const float* fe_w2  = (const float*)d_in[14];
  const float* fe_b2  = (const float*)d_in[15];
  const float* s1_wl  = (const float*)d_in[16];
  const float* s1_bl  = (const float*)d_in[17];
  const float* s1_wr  = (const float*)d_in[18];
  const float* bn1_g  = (const float*)d_in[19];
  const float* bn1_b  = (const float*)d_in[20];
  const float* s2_wl  = (const float*)d_in[21];
  const float* s2_bl  = (const float*)d_in[22];
  const float* s2_wr  = (const float*)d_in[23];
  const float* bn2_g  = (const float*)d_in[24];
  const float* bn2_b  = (const float*)d_in[25];
  const float* s3_wl  = (const float*)d_in[26];
  const float* s3_bl  = (const float*)d_in[27];
  const float* s3_wr  = (const float*)d_in[28];
  const float* bn3_g  = (const float*)d_in[29];
  const float* bn3_b  = (const float*)d_in[30];
  const float* svi_w1 = (const float*)d_in[31];
  const float* svi_b1 = (const float*)d_in[32];
  const float* svi_w2 = (const float*)d_in[33];
  const float* svi_b2 = (const float*)d_in[34];

  const int N = in_sizes[0] / 32;
  const int E = in_sizes[2] / 2;
  const int* src = edge;
  const int* dst = edge + E;
  const int tiles = CDIV(N, 16);
  float* out = (float*)d_out;

  // workspace carve-out
  size_t off = 0;
  char* ws = (char*)d_ws;
  auto take = [&](size_t bytes) -> void* {
    void* p = ws + off;
    off += (bytes + 255) & ~(size_t)255;
    return p;
  };
  bf16*  B0   = (bf16*)take((size_t)N * 64 * sizeof(bf16));
  bf16*  B1   = (bf16*)take((size_t)N * 64 * sizeof(bf16));
  float* S    = (float*)take((size_t)N * 64 * sizeof(float));
  float* P    = (float*)take((size_t)N * 64 * sizeof(float));
  float* CNT  = (float*)take((size_t)N * sizeof(float));
  float* STAT = (float*)take(256 * sizeof(float));
  // bf16 copies of the WMMA weights (converted once per launch)
  bf16* w_fe1 = (bf16*)take(64 * 32 * sizeof(bf16));
  bf16* w_fe2 = (bf16*)take(64 * 64 * sizeof(bf16));
  bf16* w1l   = (bf16*)take(64 * 64 * sizeof(bf16));
  bf16* w1r   = (bf16*)take(64 * 64 * sizeof(bf16));
  bf16* w2l   = (bf16*)take(64 * 64 * sizeof(bf16));
  bf16* w2r   = (bf16*)take(64 * 64 * sizeof(bf16));
  bf16* w3l   = (bf16*)take(32 * 64 * sizeof(bf16));
  bf16* w3r   = (bf16*)take(32 * 64 * sizeof(bf16));

  const int NB = CDIV(N, 256);
  const int TB = CDIV(tiles, 4);   // 4 waves / 128-thread block
  const int EL64 = N * 64;

  // weight conversion fp32 -> bf16 (tiny, once per launch)
  cvt_w_bf16<<<CDIV(64 * 32, 256), 256, 0, stream>>>(fe_w1, w_fe1, 64 * 32);
  cvt_w_bf16<<<CDIV(64 * 64, 256), 256, 0, stream>>>(fe_w2, w_fe2, 64 * 64);
  cvt_w_bf16<<<CDIV(64 * 64, 256), 256, 0, stream>>>(s1_wl, w1l, 64 * 64);
  cvt_w_bf16<<<CDIV(64 * 64, 256), 256, 0, stream>>>(s1_wr, w1r, 64 * 64);
  cvt_w_bf16<<<CDIV(64 * 64, 256), 256, 0, stream>>>(s2_wl, w2l, 64 * 64);
  cvt_w_bf16<<<CDIV(64 * 64, 256), 256, 0, stream>>>(s2_wr, w2r, 64 * 64);
  cvt_w_bf16<<<CDIV(32 * 64, 256), 256, 0, stream>>>(s3_wl, w3l, 32 * 64);
  cvt_w_bf16<<<CDIV(32 * 64, 256), 256, 0, stream>>>(s3_wr, w3r, 32 * 64);

  // degrees (once; shared by all 3 SAGE layers)
  zero_f32<<<NB, 256, 0, stream>>>(CNT, N);
  degree_kernel<<<CDIV(E, 256), 256, 0, stream>>>(dst, CNT, E);

  // context gate + softmax + layernorm -> x0 bf16 [N,32]
  ctx_gate_ln<<<NB, 256, 0, stream>>>(acc, ctx, ctx_w1, ctx_b1, ctx_w2, ctx_b2,
                                      attn_w, attn_b, base_i, ln_g, ln_b, B0, N);

  // feature encoder (WMMA): [N,32]->[N,64]->[N,64]
  gemm_bias_relu<32, 64><<<TB, 128, 0, stream>>>(B0, w_fe1, fe_b1, B1, N);
  gemm_bias_relu<64, 64><<<TB, 128, 0, stream>>>(B1, w_fe2, fe_b2, B0, N);

  // ----- SAGE layer 1 (in B0 -> out B1) -----
  zero_f32<<<CDIV(EL64, 256), 256, 0, stream>>>(S, EL64);
  aggregate<64><<<CDIV(E * 4, 256), 256, 0, stream>>>(B0, src, dst, S, E);
  sage_combine<64, 64><<<TB, 128, 0, stream>>>(B0, S, CNT, w1l, s1_bl, w1r, P, N);
  zero_f32<<<1, 256, 0, stream>>>(STAT, 128);
  bn_stats<64><<<512, 256, 0, stream>>>(P, STAT, N);
  bn_apply_relu<64><<<CDIV(EL64, 256), 256, 0, stream>>>(P, STAT, bn1_g, bn1_b, B1, N);

  // ----- SAGE layer 2 (in B1 -> out B0) -----
  zero_f32<<<CDIV(EL64, 256), 256, 0, stream>>>(S, EL64);
  aggregate<64><<<CDIV(E * 4, 256), 256, 0, stream>>>(B1, src, dst, S, E);
  sage_combine<64, 64><<<TB, 128, 0, stream>>>(B1, S, CNT, w2l, s2_bl, w2r, P, N);
  zero_f32<<<1, 256, 0, stream>>>(STAT, 128);
  bn_stats<64><<<512, 256, 0, stream>>>(P, STAT, N);
  bn_apply_relu<64><<<CDIV(EL64, 256), 256, 0, stream>>>(P, STAT, bn2_g, bn2_b, B0, N);

  // ----- SAGE layer 3 (in B0 [N,64] -> out B1 [N,32]) -----
  zero_f32<<<CDIV(EL64, 256), 256, 0, stream>>>(S, EL64);
  aggregate<64><<<CDIV(E * 4, 256), 256, 0, stream>>>(B0, src, dst, S, E);
  sage_combine<64, 32><<<TB, 128, 0, stream>>>(B0, S, CNT, w3l, s3_bl, w3r, P, N);
  zero_f32<<<1, 256, 0, stream>>>(STAT, 64);
  bn_stats<32><<<512, 256, 0, stream>>>(P, STAT, N);
  bn_apply_relu<32><<<CDIV(N * 32, 256), 256, 0, stream>>>(P, STAT, bn3_g, bn3_b, B1, N);

  // SVI head -> f32 predictions
  svi_head<<<NB, 256, 0, stream>>>(B1, svi_w1, svi_b1, svi_w2, svi_b2, out, N);
}
